// CNN_GRU_26920855011809
// MI455X (gfx1250) — matrix operand order repaired and checked by reading, not compiled, not run on hardware
//
#include <hip/hip_runtime.h>

#define B_    512
#define T_    96
#define PL    48
#define CIN   7
#define D_    128
#define KS_   5
#define LFULL (T_ + PL)   // 144

typedef __attribute__((ext_vector_type(16))) _Float16 v16h;
typedef __attribute__((ext_vector_type(8)))  _Float16 v8h;
typedef __attribute__((ext_vector_type(8)))  float    v8f;
typedef __attribute__((ext_vector_type(4)))  float    v4f;

// ---------------------------------------------------------------------------
// WMMA fragment helpers (v_wmma_f32_16x16x32_f16, wave32)
// A (16x32 f16): lane m=lane&15, hh=lane>>4. elements 0..15 = K[16*hh .. 16*hh+15]
//                (row-major source)
// B (32x16 f16): lane n=lane&15, kh=lane>>4. elements 0..15 = K[16*kh .. +15]
//                for column n (source stored [n][k] row-major, k contiguous)
// C/D (16x16 f32): vgpr r -> (M = r + 8*hh, N = lane&15)
// ---------------------------------------------------------------------------
__device__ __forceinline__ v16h a_frag_lds(const _Float16* rowp, int k0, int hh) {
    v16h a;
    const v8h lo = *(const v8h*)(rowp + k0 + 16 * hh);
    const v8h hi = *(const v8h*)(rowp + k0 + 16 * hh + 8);
#pragma unroll
    for (int j = 0; j < 8; ++j) { a[j] = lo[j]; a[8 + j] = hi[j]; }
    return a;
}

__device__ __forceinline__ v16h b_frag(const _Float16* __restrict__ W,
                                       int ldb, int ncol, int kh, int k0) {
    return *(const v16h*)(W + (size_t)ncol * ldb + k0 + kh * 16);
}

// ---------------------------------------------------------------------------
// One-time prep kernels
// ---------------------------------------------------------------------------
__global__ void k_temb(const int* __restrict__ ym,
                       const float* __restrict__ he, const float* __restrict__ we,
                       const float* __restrict__ de, const float* __restrict__ me,
                       float* __restrict__ temb) {
    int idx = blockIdx.x * blockDim.x + threadIdx.x;   // B * LFULL * D
    if (idx >= B_ * LFULL * D_) return;
    int d  = idx & (D_ - 1);
    int bl = idx >> 7;
    const int* y = ym + (size_t)bl * 4;
    temb[idx] = he[y[0] * D_ + d] + we[y[1] * D_ + d] +
                de[y[2] * D_ + d] + me[y[3] * D_ + d];
}

__global__ void k_init_xcat(const float* __restrict__ xe, float* __restrict__ xcat) {
    int idx = blockIdx.x * blockDim.x + threadIdx.x;   // B * LFULL * CIN
    if (idx >= B_ * LFULL * CIN) return;
    int c  = idx % CIN;
    int bl = idx / CIN;
    int b  = bl / LFULL;
    int l  = bl % LFULL;
    xcat[idx] = (l < T_) ? xe[((size_t)b * T_ + l) * CIN + c] : 0.f;
}

__global__ void k_prep_w(const float* __restrict__ c1, const float* __restrict__ c2,
                         const float* __restrict__ c3,
                         const float* __restrict__ wi, const float* __restrict__ wh,
                         _Float16* __restrict__ wt1, _Float16* __restrict__ wt2,
                         _Float16* __restrict__ wt3,
                         _Float16* __restrict__ wi16, _Float16* __restrict__ wh16) {
    int idx = blockIdx.x * blockDim.x + threadIdx.x;
    const int NC = D_ * D_ * KS_;                      // 81920
    if (idx < NC) {                                    // [O][I][K] -> [K][O][I]
        int k   = idx % KS_;
        int oi  = idx / KS_;
        int iin = oi % D_;
        int o   = oi / D_;
        int dst = (k * D_ + o) * D_ + iin;
        wt1[dst] = (_Float16)c1[idx];
        wt2[dst] = (_Float16)c2[idx];
        wt3[dst] = (_Float16)c3[idx];
    }
    if (idx < 3 * D_ * D_) {                           // 49152
        wi16[idx] = (_Float16)wi[idx];
        wh16[idx] = (_Float16)wh[idx];
    }
}

// ---------------------------------------------------------------------------
// Per-step: value embedding + temporal embedding -> f16 activations [B*T][D]
// ---------------------------------------------------------------------------
__global__ void k_embed(const float* __restrict__ xcat,
                        const float* __restrict__ Wv, const float* __restrict__ bv,
                        const float* __restrict__ temb,
                        _Float16* __restrict__ xe, int i) {
    int idx = blockIdx.x * blockDim.x + threadIdx.x;   // B * T * D
    if (idx >= B_ * T_ * D_) return;
    int d   = idx & (D_ - 1);
    int row = idx >> 7;                                // b*T + t
    int b   = row / T_;
    int t   = row - b * T_;
    int l   = i + t;
    const float* xc = xcat + ((size_t)b * LFULL + l) * CIN;
    float acc = bv[d] + temb[((size_t)b * LFULL + l) * D_ + d];
#pragma unroll
    for (int c = 0; c < CIN; ++c) acc += xc[c] * Wv[d * CIN + c];
    xe[idx] = (_Float16)acc;
}

// ---------------------------------------------------------------------------
// Per-step: Conv1d(D,D,5,pad=2) + ReLU as 5 shifted WMMA GEMM taps.
// grid = B*(T/16) blocks, 256 thr (8 waves); wave w -> output column tile w*16.
// Input tile (20 rows, zero-padded halo) staged once in LDS -> branch-free
// ds_load_b128 fragment gathers; output repacked via LDS -> coalesced b128.
// ---------------------------------------------------------------------------
__global__ __launch_bounds__(256) void k_conv(const _Float16* __restrict__ X,
                                              const _Float16* __restrict__ Wt, // [K][O][I] f16
                                              const float* __restrict__ bias,
                                              _Float16* __restrict__ Y) {
    __shared__ __align__(32) _Float16 xs[16 + KS_ - 1][D_];   // 20 x 128
    __shared__ __align__(32) _Float16 ys[16][D_];

    const int tid  = threadIdx.x;
    const int lane = tid & 31;
    const int wave = tid >> 5;
    const int m    = lane & 15;
    const int hh   = lane >> 4;
    const int rt   = blockIdx.x;
    const int b    = rt / (T_ / 16);
    const int t0   = (rt % (T_ / 16)) * 16;
    const int n0   = wave * 16;

    // hint: pull this wave's weight columns toward the caches
    __builtin_prefetch(Wt + (size_t)n0 * D_, 0, 3);

    // stage 20 input rows (zero halo) : 20 rows x 16 v8h chunks = 320 chunks
    for (int e = tid; e < 20 * 16; e += 256) {
        const int rr = e >> 4;
        const int ch = e & 15;
        const int t  = t0 + rr - 2;
        v8h v;
        if (t >= 0 && t < T_) {
            v = *(const v8h*)(X + (size_t)(b * T_ + t) * D_ + ch * 8);
        } else {
#pragma unroll
            for (int j = 0; j < 8; ++j) v[j] = (_Float16)0.f;
        }
        *(v8h*)&xs[rr][ch * 8] = v;
    }
    __syncthreads();

    v8f c;
    const float bvv = bias[n0 + m];
#pragma unroll
    for (int r = 0; r < 8; ++r) c[r] = bvv;

#pragma unroll
    for (int k = 0; k < KS_; ++k) {
        const _Float16* rowp = &xs[m + k][0];          // t = t0 + m + k - 2
        const _Float16* Wk   = Wt + (size_t)k * D_ * D_;
#pragma unroll
        for (int kc = 0; kc < 4; ++kc) {
            v16h a  = a_frag_lds(rowp, kc * 32, hh);
            v16h bm = b_frag(Wk, D_, n0 + m, hh, kc * 32);
            c = __builtin_amdgcn_wmma_f32_16x16x32_f16(false, a, false, bm,
                                                       (short)0, c, false, false);
        }
    }

    // ReLU -> repack through LDS -> coalesced stores
#pragma unroll
    for (int r = 0; r < 8; ++r) {
        float v = c[r];
        v = v > 0.f ? v : 0.f;
        ys[r + 8 * hh][n0 + m] = (_Float16)v;
    }
    __syncthreads();
    {
        const int rr = tid >> 4;                       // 256 = 16 rows x 16 chunks
        const int ch = tid & 15;
        *(v8h*)(Y + (size_t)(b * T_ + t0 + rr) * D_ + ch * 8) =
            *(const v8h*)&ys[rr][ch * 8];
    }
}

// ---------------------------------------------------------------------------
// Per-step: gx = x @ Wi^T + bi  -> f32 [B*T][384]
// grid = (B*T)/16 blocks, 8 waves; wave handles 3 column tiles (24 total).
// A tile staged once in LDS and shared by all waves; output repacked via LDS.
// ---------------------------------------------------------------------------
__global__ __launch_bounds__(256) void k_gx(const _Float16* __restrict__ X,
                                            const _Float16* __restrict__ Wi,
                                            const float* __restrict__ bi,
                                            float* __restrict__ GX) {
    __shared__ __align__(32) _Float16 xs[16][D_];
    __shared__ __align__(32) float    ys[16][3 * D_];

    const int tid  = threadIdx.x;
    const int lane = tid & 31;
    const int wave = tid >> 5;
    const int m    = lane & 15;
    const int hh   = lane >> 4;
    const int row0 = blockIdx.x * 16;

    __builtin_prefetch(Wi + (size_t)(wave * 16) * D_, 0, 3);

    {   // stage 16 rows: exactly 256 v8h chunks
        const int rr = tid >> 4;
        const int ch = tid & 15;
        *(v8h*)&xs[rr][ch * 8] =
            *(const v8h*)(X + (size_t)(row0 + rr) * D_ + ch * 8);
    }
    __syncthreads();

    v16h afr[4];
#pragma unroll
    for (int kc = 0; kc < 4; ++kc) afr[kc] = a_frag_lds(&xs[m][0], kc * 32, hh);

#pragma unroll
    for (int cg = 0; cg < 3; ++cg) {
        const int n0 = (wave + cg * 8) * 16;           // 0..368
        v8f c;
        const float bvv = bi[n0 + m];
#pragma unroll
        for (int r = 0; r < 8; ++r) c[r] = bvv;
#pragma unroll
        for (int kc = 0; kc < 4; ++kc) {
            v16h bm = b_frag(Wi, D_, n0 + m, hh, kc * 32);
            c = __builtin_amdgcn_wmma_f32_16x16x32_f16(false, afr[kc], false, bm,
                                                       (short)0, c, false, false);
        }
#pragma unroll
        for (int r = 0; r < 8; ++r) ys[r + 8 * hh][n0 + m] = c[r];
    }
    __syncthreads();

    // 16 rows x 384 f32 = 1536 v4f chunks; 6 per thread
    for (int e = tid; e < 16 * 96; e += 256) {
        const int rr = e / 96;
        const int ch = e % 96;
        *(v4f*)(GX + (size_t)(row0 + rr) * (3 * D_) + ch * 4) =
            *(const v4f*)&ys[rr][ch * 4];
    }
}

// ---------------------------------------------------------------------------
// Per-step: GRU scan over T=96 + FC(128->7) + write prediction into x_cat.
// grid = B/16 blocks; h kept in LDS; per timestep gh = h @ Wh^T via WMMA
// (A fragments straight from LDS), then gates in VALU.
// ---------------------------------------------------------------------------
__global__ __launch_bounds__(256) void k_gru(const float* __restrict__ GX,
                                             const _Float16* __restrict__ Wh,
                                             const float* __restrict__ bh,
                                             const float* __restrict__ fcw,
                                             const float* __restrict__ fcb,
                                             float* __restrict__ xcat, int i) {
    __shared__ __align__(32) float    hs [16][D_];
    __shared__ __align__(32) _Float16 h16[16][D_];
    __shared__ __align__(32) float    ghs[16][3 * D_];

    const int tid  = threadIdx.x;
    const int lane = tid & 31;
    const int wave = tid >> 5;
    const int m    = lane & 15;
    const int hh   = lane >> 4;
    const int b0   = blockIdx.x * 16;

    for (int e = tid; e < 16 * D_; e += 256) (&hs[0][0])[e] = 0.f;
    __syncthreads();

    for (int t = 0; t < T_; ++t) {
        for (int e = tid; e < 16 * D_; e += 256)
            (&h16[0][0])[e] = (_Float16)(&hs[0][0])[e];
        __syncthreads();

#pragma unroll
        for (int cg = 0; cg < 3; ++cg) {
            const int n0 = (wave + cg * 8) * 16;
            v8f c;
            const float bvv = bh[n0 + m];
#pragma unroll
            for (int r = 0; r < 8; ++r) c[r] = bvv;
#pragma unroll
            for (int kc = 0; kc < 4; ++kc) {
                v16h a  = a_frag_lds(&h16[m][0], kc * 32, hh);
                v16h bm = b_frag(Wh, D_, n0 + m, hh, kc * 32);
                c = __builtin_amdgcn_wmma_f32_16x16x32_f16(false, a, false, bm,
                                                           (short)0, c, false, false);
            }
#pragma unroll
            for (int r = 0; r < 8; ++r) ghs[r + 8 * hh][n0 + m] = c[r];
        }
        __syncthreads();

        for (int e = tid; e < 16 * D_; e += 256) {
            const int r = e >> 7;
            const int d = e & (D_ - 1);
            const float* gxr = GX + ((size_t)(b0 + r) * T_ + t) * (3 * D_);
            const float gr = gxr[d]          + ghs[r][d];
            const float gz = gxr[D_ + d]     + ghs[r][D_ + d];
            const float rr = 1.f / (1.f + __expf(-gr));
            const float zz = 1.f / (1.f + __expf(-gz));
            const float gn = gxr[2 * D_ + d] + rr * ghs[r][2 * D_ + d];
            const float nn = tanhf(gn);
            hs[r][d] = (1.f - zz) * nn + zz * hs[r][d];
        }
        __syncthreads();
    }

    if (tid < 16 * CIN) {
        const int r = tid / CIN;
        const int c = tid % CIN;
        float acc = fcb[c];
#pragma unroll 8
        for (int d = 0; d < D_; ++d) acc += hs[r][d] * fcw[c * D_ + d];
        xcat[((size_t)(b0 + r) * LFULL + T_ + i) * CIN + c] = acc;
    }
}

// ---------------------------------------------------------------------------
// Final: copy predictions out of x_cat.
// ---------------------------------------------------------------------------
__global__ void k_out(const float* __restrict__ xcat, float* __restrict__ out) {
    int idx = blockIdx.x * blockDim.x + threadIdx.x;   // B * PL * CIN
    if (idx >= B_ * PL * CIN) return;
    int c  = idx % CIN;
    int bt = idx / CIN;
    int b  = bt / PL;
    int t  = bt % PL;
    out[idx] = xcat[((size_t)b * LFULL + T_ + t) * CIN + c];
}

// ---------------------------------------------------------------------------
extern "C" void kernel_launch(void* const* d_in, const int* in_sizes, int n_in,
                              void* d_out, int out_size, void* d_ws, size_t ws_size,
                              hipStream_t stream) {
    (void)in_sizes; (void)n_in; (void)out_size; (void)ws_size;

    const float* x_enc   = (const float*)d_in[0];
    const int*   y_mark  = (const int*)  d_in[2];
    const float* hour_e  = (const float*)d_in[3];
    const float* week_e  = (const float*)d_in[4];
    const float* day_e   = (const float*)d_in[5];
    const float* mon_e   = (const float*)d_in[6];
    const float* W_val   = (const float*)d_in[7];
    const float* b_val   = (const float*)d_in[8];
    const float* c1w     = (const float*)d_in[9];
    const float* c1b     = (const float*)d_in[10];
    const float* c2w     = (const float*)d_in[11];
    const float* c2b     = (const float*)d_in[12];
    const float* c3w     = (const float*)d_in[13];
    const float* c3b     = (const float*)d_in[14];
    const float* gru_Wi  = (const float*)d_in[15];
    const float* gru_Wh  = (const float*)d_in[16];
    const float* gru_bi  = (const float*)d_in[17];
    const float* gru_bh  = (const float*)d_in[18];
    const float* fc_w    = (const float*)d_in[19];
    const float* fc_b    = (const float*)d_in[20];

    char* ws = (char*)d_ws;
    size_t off = 0;
    auto alloc = [&](size_t bytes) -> char* {
        char* p = ws + off;
        off = (off + bytes + 255) & ~(size_t)255;
        return p;
    };

    const size_t NROW = (size_t)B_ * T_;               // 49152
    float*    temb  = (float*)   alloc((size_t)B_ * LFULL * D_ * 4);
    float*    xcat  = (float*)   alloc((size_t)B_ * LFULL * CIN * 4);
    _Float16* xe    = (_Float16*)alloc(NROW * D_ * 2);
    _Float16* bufA  = (_Float16*)alloc(NROW * D_ * 2);
    _Float16* bufB  = (_Float16*)alloc(NROW * D_ * 2);
    float*    gx    = (float*)   alloc(NROW * (3 * D_) * 4);
    _Float16* wt1   = (_Float16*)alloc((size_t)KS_ * D_ * D_ * 2);
    _Float16* wt2   = (_Float16*)alloc((size_t)KS_ * D_ * D_ * 2);
    _Float16* wt3   = (_Float16*)alloc((size_t)KS_ * D_ * D_ * 2);
    _Float16* wi16  = (_Float16*)alloc((size_t)3 * D_ * D_ * 2);
    _Float16* wh16  = (_Float16*)alloc((size_t)3 * D_ * D_ * 2);

    // one-time prep
    k_temb<<<(B_ * LFULL * D_ + 255) / 256, 256, 0, stream>>>(
        y_mark, hour_e, week_e, day_e, mon_e, temb);
    k_init_xcat<<<(B_ * LFULL * CIN + 255) / 256, 256, 0, stream>>>(x_enc, xcat);
    k_prep_w<<<(D_ * D_ * KS_ + 255) / 256, 256, 0, stream>>>(
        c1w, c2w, c3w, gru_Wi, gru_Wh, wt1, wt2, wt3, wi16, wh16);

    const int rowTiles = B_ * (T_ / 16);               // 3072

    for (int i = 0; i < PL; ++i) {
        k_embed<<<(B_ * T_ * D_ + 255) / 256, 256, 0, stream>>>(
            xcat, W_val, b_val, temb, xe, i);
        k_conv<<<rowTiles, 256, 0, stream>>>(xe,   wt1, c1b, bufA);
        k_conv<<<rowTiles, 256, 0, stream>>>(bufA, wt2, c2b, bufB);
        k_conv<<<rowTiles, 256, 0, stream>>>(bufB, wt3, c3b, bufA);
        k_gx  <<<rowTiles, 256, 0, stream>>>(bufA, wi16, gru_bi, gx);
        k_gru <<<B_ / 16, 256, 0, stream>>>(gx, wh16, gru_bh, fc_w, fc_b, xcat, i);
    }

    k_out<<<(B_ * PL * CIN + 255) / 256, 256, 0, stream>>>(xcat, (float*)d_out);
}
